// QKVParallelLinearWithToppings_63926293234284
// MI455X (gfx1250) — compile-verified, matrix operand order checked by reading, not run on hardware
//
#include <hip/hip_runtime.h>
#include <hip/hip_bf16.h>

typedef __attribute__((ext_vector_type(16))) __bf16 v16bf;
typedef __attribute__((ext_vector_type(8)))  float  v8f;
typedef int v4i_g __attribute__((vector_size(16)));   // async-to-LDS builtin param
typedef unsigned int u32x4 __attribute__((ext_vector_type(4)));
typedef int i32x8 __attribute__((ext_vector_type(8)));
typedef int i32x4 __attribute__((ext_vector_type(4)));

#define S_TOK 2048
#define HDIM  2048
#define NOUT  6144
#define DQCOL 2048
#define NKV   4096   // 2*DKV
#define LADP  4
#define NGRP  16     // HDIM / 128
#define RNK   32

#ifndef __has_builtin
#define __has_builtin(x) 0
#endif

#if __has_builtin(__builtin_amdgcn_tensor_load_to_lds) && __has_builtin(__builtin_amdgcn_s_wait_tensorcnt)
#define USE_TDM 1
#else
#define USE_TDM 0
#endif

#if __has_builtin(__builtin_amdgcn_global_load_async_to_lds_b128)
#define USE_ASYNC 1
#else
#define USE_ASYNC 0
#endif

#if __has_builtin(__builtin_amdgcn_s_wait_asynccnt)
#define WAIT_ASYNC(n) __builtin_amdgcn_s_wait_asynccnt(n)
#else
#define WAIT_ASYNC(n) asm volatile("s_wait_asynccnt %0" ::"i"(n) : "memory")
#endif

// address-space-qualified pointers (typed pointee: AS attr on void gets dropped)
#define ASG_V4(p) ((__attribute__((address_space(1))) v4i_g*)(p))
#define ASL_V4(p) ((__attribute__((address_space(3))) v4i_g*)(p))
#define LDSOFF(p) ((unsigned)(unsigned long long)(__attribute__((address_space(3))) char*)(p))

static __device__ __forceinline__ unsigned short f2bfu(float f) {
  unsigned u = __builtin_bit_cast(unsigned, f);
  u += 0x7fffu + ((u >> 16) & 1u);          // round-to-nearest-even
  return (unsigned short)(u >> 16);
}
static __device__ __forceinline__ __bf16 f2bf(float f) {
  unsigned short h = f2bfu(f);
  return __builtin_bit_cast(__bf16, h);
}
// two f32 -> packed 2xbf16 dword (v_cvt_pk_bf16_f32 when available)
static __device__ __forceinline__ unsigned pk2bf(float a, float b) {
#if __has_builtin(__builtin_amdgcn_cvt_pk_bf16_f32)
  typedef __attribute__((ext_vector_type(2))) __bf16 v2bf;
  v2bf p = __builtin_amdgcn_cvt_pk_bf16_f32(a, b);
  return __builtin_bit_cast(unsigned, p);
#else
  return (unsigned)f2bfu(a) | ((unsigned)f2bfu(b) << 16);
#endif
}

#if USE_TDM
// Program one 2-D TDM descriptor (D#, ISA 8.3/8.4): tile_w x tile_h elements of
// 4 bytes from a row-major tensor with pitch stride_w, into contiguous LDS.
// 6-arg builtin form: (g0 u32x4, g1 i32x8, g2 i32x4, g3 i32x4, extra i32x8, cpol)
static __device__ __forceinline__ void tdm_load_2d(const float* gbase, unsigned ldsoff,
                                                   unsigned tensor_w, unsigned tensor_h,
                                                   unsigned stride_w,
                                                   unsigned tile_w, unsigned tile_h) {
  unsigned long long ga = (unsigned long long)(size_t)gbase;
  u32x4 g0;
  g0[0] = 1u;                                          // count=1, user descriptor
  g0[1] = ldsoff;                                      // lds_addr (bytes)
  g0[2] = (unsigned)(ga & 0xffffffffu);                // global_addr[31:0]
  g0[3] = (unsigned)((ga >> 32) & 0x01ffffffu) | (2u << 30);  // addr[56:32], type=2
  i32x8 g1;
  g1[0] = (int)(2u << 16);                             // data_size=2 (4 bytes)
  g1[1] = (int)(tensor_w << 16);                       // tensor_dim0[15:0]
  g1[2] = (int)((tensor_w >> 16) | (tensor_h << 16));  // dim0 hi | dim1 lo
  g1[3] = (int)((tensor_h >> 16) | (tile_w << 16));    // dim1 hi | tile_dim0
  g1[4] = (int)(tile_h & 0xffffu);                     // tile_dim1 (tile_dim2=0)
  g1[5] = (int)stride_w;                               // tensor_dim0_stride[31:0]
  g1[6] = 0;
  g1[7] = 0;
  i32x4 z4 = (i32x4){0, 0, 0, 0};
  i32x8 z8 = (i32x8){0, 0, 0, 0, 0, 0, 0, 0};
  __builtin_amdgcn_tensor_load_to_lds(g0, g1, z4, z4, z8, 0);
}
#endif

// ---------------------------------------------------------------------------
// Kernel 1: U[t, 0:64] = x[t,:] @ A_buffer_qkv[idx_t, :, 0:64]   (bf16 out)
// ---------------------------------------------------------------------------
__global__ void __launch_bounds__(64)
lora_u_kernel(const float* __restrict__ x,
              const float* __restrict__ A,
              const int* __restrict__ idx,
              unsigned short* __restrict__ U) {
  const int t = blockIdx.x;
  const int j = threadIdx.x;                 // 0..63
  const int l = idx[t];
  const float* a  = A + ((size_t)l * HDIM) * (2 * RNK) + j;
  const float* xr = x + (size_t)t * HDIM;
  float sum = 0.f;
#pragma unroll 8
  for (int h = 0; h < HDIM; ++h) sum += xr[h] * a[(size_t)h * (2 * RNK)];
  U[(size_t)t * (2 * RNK) + j] = f2bfu(sum);
}

// ---------------------------------------------------------------------------
// Kernel 2: fused  out = x@W_base + sum_l mask_l(x)@dequant(dW_l) + mask_l(U)@B_l
// Block: 256 threads (8 waves), tile 256 rows x 64 cols; waves 4 row x 2 col.
// All global->LDS movement via TDM descriptors (TENSORcnt), double-buffered.
// ---------------------------------------------------------------------------
__global__ void __launch_bounds__(256)
qkv_toppings_kernel(const float* __restrict__ x,
                    const float* __restrict__ Wb,
                    const float* __restrict__ Dq,
                    const float* __restrict__ Mq,
                    const float* __restrict__ Sq,
                    const float* __restrict__ Dkv,
                    const float* __restrict__ Mkv,
                    const float* __restrict__ Skv,
                    const float* __restrict__ Bq,
                    const float* __restrict__ Bkv,
                    const int*   __restrict__ idx,
                    const unsigned short* __restrict__ U,
                    float* __restrict__ out) {
#if USE_TDM
  __shared__ float rawX[2][256 * 32];            // raw f32 x tiles (TDM dest)
  __shared__ float rawW[2][5][32 * 64];          // raw f32 weight tiles (TDM dest)
#elif USE_ASYNC
  __shared__ float rawX[2][256 * 32];            // raw f32 x tiles (async dest)
#endif
  __shared__ unsigned short xs[256 * 32];        // x tile, bf16, [row][k]
  __shared__ unsigned short wl[5 * 64 * 32];     // 5 weight tiles, bf16, [mat][col][k]

  const int tid     = threadIdx.x;
  const int lane    = tid & 31;
  const int wave    = tid >> 5;
  const int waveRow = wave & 3;                  // 4 groups of 64 rows
  const int waveCol = wave >> 2;                 // 2 groups of 32 cols
  const int mn      = lane & 15;                 // A row / B col within tile
  const int kg      = lane >> 4;                 // K half-select
  const int kb      = kg * 16;                   // K base within 32-step

  const int colBase  = blockIdx.x * 64;          // 96 col blocks
  const int rowBase0 = blockIdx.y * 256;         // 8 row blocks

  // staging coordinates for weight tiles: one column, 8 consecutive k per thread
  const int wc  = tid & 63;
  const int wk0 = (tid >> 6) * 8;

  int ridx[4];
#pragma unroll
  for (int rt = 0; rt < 4; ++rt)
    ridx[rt] = idx[rowBase0 + waveRow * 64 + rt * 16 + mn];

  v8f acc[8];
#pragma unroll
  for (int i = 0; i < 8; ++i) acc[i] = (v8f){};
  const v16bf zf = {};

#if USE_TDM
  // one wave programs the DMA: 6 descriptors per K-step (x + 5 weight tiles)
  const int inKV = (colBase >= DQCOL);
  const int cOff = inKV ? (colBase - DQCOL) : colBase;
  const int ldd  = inKV ? NKV : DQCOL;
  const float* dBase = inKV ? Dkv : Dq;
  if (wave == 0) {
    tdm_load_2d(x + (size_t)rowBase0 * HDIM, LDSOFF(&rawX[0][0]),
                HDIM, S_TOK, HDIM, 32, 256);
    tdm_load_2d(Wb + (size_t)0 * NOUT + colBase, LDSOFF(&rawW[0][0][0]),
                NOUT, HDIM, NOUT, 64, 32);
#pragma unroll
    for (int l = 0; l < LADP; ++l)
      tdm_load_2d(dBase + ((size_t)l * HDIM) * ldd + cOff, LDSOFF(&rawW[0][1 + l][0]),
                  ldd, HDIM, ldd, 64, 32);
  }
#elif USE_ASYNC
#pragma unroll
  for (int i = 0; i < 8; ++i) {
    int q = tid + i * 256;                       // 2048 16B chunks
    int row = q >> 3, seg = (q & 7) * 4;
    const float* gsrc = x + (size_t)(rowBase0 + row) * HDIM + seg;
    __builtin_amdgcn_global_load_async_to_lds_b128(
        ASG_V4(gsrc), ASL_V4(&rawX[0][row * 32 + seg]), 0, 0);
  }
#endif

  for (int k0 = 0; k0 < HDIM; k0 += 32) {
    const int cb = (k0 >> 5) & 1;
    (void)cb;
#if USE_TDM
    if (wave == 0) {
      const int kn = k0 + 32;
      if (kn < HDIM) {
        const int nb = cb ^ 1;
        tdm_load_2d(x + (size_t)rowBase0 * HDIM + kn, LDSOFF(&rawX[nb][0]),
                    HDIM, S_TOK, HDIM, 32, 256);
        tdm_load_2d(Wb + (size_t)kn * NOUT + colBase, LDSOFF(&rawW[nb][0][0]),
                    NOUT, HDIM, NOUT, 64, 32);
#pragma unroll
        for (int l = 0; l < LADP; ++l)
          tdm_load_2d(dBase + ((size_t)l * HDIM + kn) * ldd + cOff,
                      LDSOFF(&rawW[nb][1 + l][0]), ldd, HDIM, ldd, 64, 32);
        __builtin_amdgcn_s_wait_tensorcnt(6);    // current step's 6 DMAs done
      } else {
        __builtin_amdgcn_s_wait_tensorcnt(0);
      }
    }
#elif USE_ASYNC
    if (k0 + 32 < HDIM) {
#pragma unroll
      for (int i = 0; i < 8; ++i) {
        int q = tid + i * 256;
        int row = q >> 3, seg = (q & 7) * 4;
        const float* gsrc = x + (size_t)(rowBase0 + row) * HDIM + (k0 + 32) + seg;
        __builtin_amdgcn_global_load_async_to_lds_b128(
            ASG_V4(gsrc), ASL_V4(&rawX[cb ^ 1][row * 32 + seg]), 0, 0);
      }
      WAIT_ASYNC(8);
    } else {
      WAIT_ASYNC(0);
    }
#endif
    __syncthreads();                             // DMA'd tiles visible to all waves

    // ---- convert x tile to bf16 (packed pairs) ----
#pragma unroll
    for (int i = 0; i < 16; ++i) {
      int e   = tid + i * 256;                   // 4096 float2 slots
      int row = e >> 4;
      int kc2 = e & 15;
#if USE_TDM || USE_ASYNC
      const float2 v = *(const float2*)&rawX[cb][row * 32 + kc2 * 2];
#else
      const float2 v = *(const float2*)(x + (size_t)(rowBase0 + row) * HDIM + k0 + kc2 * 2);
#endif
      ((unsigned*)xs)[row * 16 + kc2] = pk2bf(v.x, v.y);
    }

    // ---- stage W_base tile (32x64 -> transposed bf16, packed stores) ----
    {
      float wv[8];
#if USE_TDM
      const float* wsrc = &rawW[cb][0][0];
#pragma unroll
      for (int i = 0; i < 8; ++i) wv[i] = wsrc[(wk0 + i) * 64 + wc];
#else
      const float* wbp = Wb + (size_t)(k0 + wk0) * NOUT + colBase + wc;
#pragma unroll
      for (int i = 0; i < 8; ++i) wv[i] = wbp[(size_t)i * NOUT];
#endif
      unsigned* dst = (unsigned*)(wl + (0 * 64 + wc) * 32 + wk0);
#pragma unroll
      for (int i = 0; i < 4; ++i) dst[i] = pk2bf(wv[2 * i], wv[2 * i + 1]);
    }
    // ---- stage dequantized delta tiles for 4 adapters ----
    {
      const int g = k0 >> 7;                     // 128-row quant group
#pragma unroll
      for (int l = 0; l < LADP; ++l) {
        const float *mp, *sp;
        if (colBase < DQCOL) {
          mp = Mq  + ((size_t)l * NGRP + g) * DQCOL + colBase + wc;
          sp = Sq  + ((size_t)l * NGRP + g) * DQCOL + colBase + wc;
        } else {
          mp = Mkv + ((size_t)l * NGRP + g) * NKV + (colBase - DQCOL) + wc;
          sp = Skv + ((size_t)l * NGRP + g) * NKV + (colBase - DQCOL) + wc;
        }
        const float mv = *mp, sv = *sp;
        float wv[8];
#if USE_TDM
        const float* wsrc = &rawW[cb][1 + l][0];
#pragma unroll
        for (int i = 0; i < 8; ++i) wv[i] = (wsrc[(wk0 + i) * 64 + wc] - mv) * sv;
#else
        const float* dp;
        if (colBase < DQCOL)
          dp = Dq  + ((size_t)l * HDIM + k0 + wk0) * DQCOL + colBase + wc;
        else
          dp = Dkv + ((size_t)l * HDIM + k0 + wk0) * NKV + (colBase - DQCOL) + wc;
        const int lddl = (colBase < DQCOL) ? DQCOL : NKV;
#pragma unroll
        for (int i = 0; i < 8; ++i) wv[i] = (dp[(size_t)i * lddl] - mv) * sv;
#endif
        unsigned* dst = (unsigned*)(wl + ((1 + l) * 64 + wc) * 32 + wk0);
#pragma unroll
        for (int i = 0; i < 4; ++i) dst[i] = pk2bf(wv[2 * i], wv[2 * i + 1]);
      }
    }
#if !USE_TDM
    if (k0 + 32 < HDIM) {
      __builtin_prefetch(Wb + (size_t)(k0 + 32 + wk0) * NOUT + colBase + wc, 0, 1);
    }
#endif
    __syncthreads();                             // staged bf16 tiles visible

    // ---- A fragments from LDS (32B contiguous per lane) ----
    v16bf af[4];
#pragma unroll
    for (int rt = 0; rt < 4; ++rt) {
      int rib = waveRow * 64 + rt * 16 + mn;
      af[rt] = *(const v16bf*)(xs + rib * 32 + kb);
    }
    // ---- base GEMM pass ----
#pragma unroll
    for (int ct = 0; ct < 2; ++ct) {
      int c = waveCol * 32 + ct * 16 + mn;
      v16bf b = *(const v16bf*)(wl + (0 * 64 + c) * 32 + kb);
#pragma unroll
      for (int rt = 0; rt < 4; ++rt)
        acc[rt * 2 + ct] = __builtin_amdgcn_wmma_f32_16x16x32_bf16(
            false, af[rt], false, b, (short)0, acc[rt * 2 + ct], false, false);
    }
    // ---- 4 masked adapter passes ----
#pragma unroll
    for (int l = 0; l < LADP; ++l) {
      v16bf am[4];
#pragma unroll
      for (int rt = 0; rt < 4; ++rt) am[rt] = (ridx[rt] == l) ? af[rt] : zf;
#pragma unroll
      for (int ct = 0; ct < 2; ++ct) {
        int c = waveCol * 32 + ct * 16 + mn;
        v16bf b = *(const v16bf*)(wl + ((1 + l) * 64 + c) * 32 + kb);
#pragma unroll
        for (int rt = 0; rt < 4; ++rt)
          acc[rt * 2 + ct] = __builtin_amdgcn_wmma_f32_16x16x32_bf16(
              false, am[rt], false, b, (short)0, acc[rt * 2 + ct], false, false);
      }
    }
    __syncthreads();                             // before next staging overwrite
  }

  // ---- LoRA: one K=32 WMMA per adapter (q and k output regions only) ----
  if (colBase < 4096) {
    const int jsel = (colBase >= DQCOL) ? 32 : 0;   // U cols 0:32 = q, 32:64 = k
    const float* bsrc; int ldb, coff;
    if (colBase < DQCOL) { bsrc = Bq;  ldb = DQCOL; coff = colBase; }
    else                 { bsrc = Bkv; ldb = NKV;   coff = colBase - DQCOL; }
#pragma unroll
    for (int l = 0; l < LADP; ++l) {
      const float* bp = bsrc + (size_t)l * RNK * ldb + coff;
#pragma unroll
      for (int ct = 0; ct < 2; ++ct) {
        int c = waveCol * 32 + ct * 16 + mn;
        v16bf b;
#pragma unroll
        for (int j = 0; j < 16; ++j) b[j] = f2bf(bp[(size_t)(kb + j) * ldb + c]);
#pragma unroll
        for (int rt = 0; rt < 4; ++rt) {
          int rg = rowBase0 + waveRow * 64 + rt * 16 + mn;
          v16bf ua = *(const v16bf*)(U + (size_t)rg * (2 * RNK) + jsel + kb);
          v16bf am = (ridx[rt] == l) ? ua : zf;
          acc[rt * 2 + ct] = __builtin_amdgcn_wmma_f32_16x16x32_bf16(
              false, am, false, b, (short)0, acc[rt * 2 + ct], false, false);
        }
      }
    }
  }

  // ---- store D tiles: VGPR v holds row (kg*8 + v), lanes 0..15 = cols ----
#pragma unroll
  for (int rt = 0; rt < 4; ++rt) {
#pragma unroll
    for (int ct = 0; ct < 2; ++ct) {
      int col = colBase + waveCol * 32 + ct * 16 + mn;
#pragma unroll
      for (int v = 0; v < 8; ++v) {
        int row = rowBase0 + waveRow * 64 + rt * 16 + kg * 8 + v;
        out[(size_t)row * NOUT + col] = acc[rt * 2 + ct][v];
      }
    }
  }
}

extern "C" void kernel_launch(void* const* d_in, const int* in_sizes, int n_in,
                              void* d_out, int out_size, void* d_ws, size_t ws_size,
                              hipStream_t stream) {
  (void)in_sizes; (void)n_in; (void)out_size; (void)ws_size;
  const float* x    = (const float*)d_in[0];
  const float* Wb   = (const float*)d_in[1];
  const float* Aq   = (const float*)d_in[2];
  const float* Bq   = (const float*)d_in[3];
  const float* Bkv  = (const float*)d_in[4];
  const float* Dq   = (const float*)d_in[5];
  const float* Mq   = (const float*)d_in[6];
  const float* Sq   = (const float*)d_in[7];
  const float* Dkv  = (const float*)d_in[8];
  const float* Mkv  = (const float*)d_in[9];
  const float* Skv  = (const float*)d_in[10];
  const int*   idx  = (const int*)d_in[11];
  float* out = (float*)d_out;
  unsigned short* U = (unsigned short*)d_ws;     // S_TOK * 64 bf16 = 256 KB

  lora_u_kernel<<<dim3(S_TOK), dim3(64), 0, stream>>>(x, Aq, idx, U);
  qkv_toppings_kernel<<<dim3(NOUT / 64, S_TOK / 256), dim3(256), 0, stream>>>(
      x, Wb, Dq, Mq, Sq, Dkv, Mkv, Skv, Bq, Bkv, idx, U, out);
}
